// P4Mamba_20392504721961
// MI455X (gfx1250) — matrix operand order repaired and checked by reading, not compiled, not run on hardware
//
#include <hip/hip_runtime.h>
#include <hip/hip_bf16.h>
#include <math.h>

// ---------------- model constants ----------------
#define B_N    4
#define L_FR   8
#define NPTS   1024
#define DIMC   512
#define DSTATE 16
#define DCONV  4
#define DTRANK 32
#define DEPTH  4
#define KNB    32
#define RADIUS_ 0.3f
#define T_N    4
#define NANCH  256
#define SEQ    1024            // T_N * NANCH
#define TOK    (B_N * SEQ)     // 4096 tokens
#define GRP    (B_N * T_N)     // 16 FPS groups

// input pytree indices (jax dict-key-sorted flattening)
#define P_BETAS  0
#define P_CONV_D 6
#define P_CONV_F 7
#define P_GENDER 8
#define P_LAYER0 14   // +0 A_log,+1 D,+2 conv_b,+3 conv_w,+4 dtW,+5 dtb,+6 inW,+7 norm_w,+8 outW,+9 xW
#define P_NORM_F 54
#define P_POS_W  55
#define P_POS_B  56
#define P_POSE   57
#define P_ROOT   63
#define P_TRANS  69
#define P_RADAR  75

typedef __bf16 bf16_t;
typedef bf16_t bf16x16 __attribute__((ext_vector_type(16)));
typedef float  f32x8   __attribute__((ext_vector_type(8)));

// ---------------- WMMA fragment loaders (CDNA5 §7.12.2 layouts) ----------------
// A 16x32 bf16: lane = 16*kh + row; elems 0..7 -> K = kh*8+0..7 ; elems 8..15 -> K = 16+kh*8+0..7
__device__ __forceinline__ bf16x16 load_frag_a(const bf16_t* __restrict__ rowp, int kh) {
    union { uint4 u[2]; bf16x16 v; } t;
    const uint4* p = reinterpret_cast<const uint4*>(rowp + kh * 8);
    t.u[0] = p[0];          // K = kh*8 .. kh*8+7
    t.u[1] = p[2];          // K = 16+kh*8 .. 16+kh*8+7   (+16 elems = +2 uint4)
    return t.v;
}
// B 32x16 bf16 (from row-major W[N,K], lane = col): elems 0..7 -> K = kh*8+0..7 ; elems 8..15 -> K = 16+kh*8+0..7
__device__ __forceinline__ bf16x16 load_frag_b(const bf16_t* __restrict__ rowp, int kh) {
    union { uint4 u[2]; bf16x16 v; } t;
    const uint4* p = reinterpret_cast<const uint4*>(rowp + kh * 8);
    t.u[0] = p[0];
    t.u[1] = p[2];
    return t.v;
}

// ---------------- GEMM: C[M,N] = A[M,K](bf16) * W[N,K]^T(bf16) (+bias) (+add) ----------------
// block = 256 (8 waves); wave -> 16 rows x 64 cols (4 wmma tiles, A-fragment reused 4x).
// Software-pipelined: fragments for step k+32 are loaded before the 4 WMMAs of step k,
// so each s_wait_loadcnt covers loads issued a full iteration earlier (L2-latency hiding
// within a single wave, on top of 8-wave multithreading).
// grid = (N/64, M/128). M multiple of 128; K multiple of 32; N multiple of 64.
__global__ void k_gemm(const bf16_t* __restrict__ A, const bf16_t* __restrict__ W,
                       const float* __restrict__ bias, const float* __restrict__ add,
                       float* __restrict__ C, int M, int N, int K) {
    int wave = threadIdx.x >> 5;
    int lane = threadIdx.x & 31;
    int r  = lane & 15;
    int kh = lane >> 4;
    int row0 = blockIdx.y * 128 + wave * 16;
    int col0 = blockIdx.x * 64;
    if (row0 >= M) return;

    const bf16_t* arow = A + (size_t)(row0 + r) * K;
    const bf16_t* w0 = W + (size_t)(col0 +  0 + r) * K;
    const bf16_t* w1 = W + (size_t)(col0 + 16 + r) * K;
    const bf16_t* w2 = W + (size_t)(col0 + 32 + r) * K;
    const bf16_t* w3 = W + (size_t)(col0 + 48 + r) * K;

    f32x8 acc[4] = {};

    // prologue: fragments for k = 0
    bf16x16 a_c  = load_frag_a(arow, kh);
    bf16x16 b_c0 = load_frag_b(w0, kh);
    bf16x16 b_c1 = load_frag_b(w1, kh);
    bf16x16 b_c2 = load_frag_b(w2, kh);
    bf16x16 b_c3 = load_frag_b(w3, kh);

    for (int k0 = 32; k0 < K; k0 += 32) {
        // prefetch next-step fragments (issued before this step's WMMAs)
        bf16x16 a_n  = load_frag_a(arow + k0, kh);
        bf16x16 b_n0 = load_frag_b(w0 + k0, kh);
        bf16x16 b_n1 = load_frag_b(w1 + k0, kh);
        bf16x16 b_n2 = load_frag_b(w2 + k0, kh);
        bf16x16 b_n3 = load_frag_b(w3 + k0, kh);
        // compute on current fragments (loads from the previous iteration)
        acc[0] = __builtin_amdgcn_wmma_f32_16x16x32_bf16(false, a_c, false, b_c0, (short)0, acc[0], false, false);
        acc[1] = __builtin_amdgcn_wmma_f32_16x16x32_bf16(false, a_c, false, b_c1, (short)0, acc[1], false, false);
        acc[2] = __builtin_amdgcn_wmma_f32_16x16x32_bf16(false, a_c, false, b_c2, (short)0, acc[2], false, false);
        acc[3] = __builtin_amdgcn_wmma_f32_16x16x32_bf16(false, a_c, false, b_c3, (short)0, acc[3], false, false);
        a_c = a_n; b_c0 = b_n0; b_c1 = b_n1; b_c2 = b_n2; b_c3 = b_n3;
    }
    // epilogue: last step
    acc[0] = __builtin_amdgcn_wmma_f32_16x16x32_bf16(false, a_c, false, b_c0, (short)0, acc[0], false, false);
    acc[1] = __builtin_amdgcn_wmma_f32_16x16x32_bf16(false, a_c, false, b_c1, (short)0, acc[1], false, false);
    acc[2] = __builtin_amdgcn_wmma_f32_16x16x32_bf16(false, a_c, false, b_c2, (short)0, acc[2], false, false);
    acc[3] = __builtin_amdgcn_wmma_f32_16x16x32_bf16(false, a_c, false, b_c3, (short)0, acc[3], false, false);

    // D layout: lanes 0-15: vgpr i -> M=i, N=lane ; lanes 16-31: vgpr i -> M=8+i, N=lane-16
    int mb = (lane < 16) ? 0 : 8;
#pragma unroll
    for (int j = 0; j < 4; ++j) {
        int col = col0 + 16 * j + r;
        float bv = bias ? bias[col] : 0.0f;
#pragma unroll
        for (int i = 0; i < 8; ++i) {
            size_t o = (size_t)(row0 + mb + i) * N + col;
            float v = acc[j][i] + bv;
            if (add) v += add[o];
            C[o] = v;
        }
    }
}

// ---------------- f32 -> bf16 conversion ----------------
__global__ void k_f2bf(const float* __restrict__ src, bf16_t* __restrict__ dst, int n) {
    int i = blockIdx.x * blockDim.x + threadIdx.x;
    if (i < n) dst[i] = (bf16_t)src[i];
}

// ---------------- FPS: one block (1024 thr) per group; dist in registers ----------------
__global__ void k_fps(const float* __restrict__ radar, float* __restrict__ anchors) {
    int g = blockIdx.x;
    int b = g / T_N, t = g % T_N;
    const float* pts = radar + (size_t)((b * L_FR + 2 * t) * NPTS) * 4;  // orig frame 2t
    __shared__ float s_rv[32];
    __shared__ int   s_ri[32];
    __shared__ int   s_idx[NANCH];
    __shared__ int   s_last;
    int tid = threadIdx.x;
    float px = pts[tid * 4 + 0], py = pts[tid * 4 + 1], pz = pts[tid * 4 + 2];
    float dmin = 1e10f;
    if (tid == 0) { s_idx[0] = 0; s_last = 0; }
    __syncthreads();
    for (int s = 1; s < NANCH; ++s) {
        int last = s_last;
        float lx = pts[last * 4 + 0], ly = pts[last * 4 + 1], lz = pts[last * 4 + 2];
        float dx = px - lx, dy = py - ly, dz = pz - lz;
        dmin = fminf(dmin, dx * dx + dy * dy + dz * dz);
        float v = dmin; int bi = tid;
#pragma unroll
        for (int m = 16; m; m >>= 1) {
            float ov = __shfl_xor(v, m, 32);
            int   oi = __shfl_xor(bi, m, 32);
            if (ov > v || (ov == v && oi < bi)) { v = ov; bi = oi; }
        }
        if ((tid & 31) == 0) { s_rv[tid >> 5] = v; s_ri[tid >> 5] = bi; }
        __syncthreads();
        if (tid < 32) {
            v = s_rv[tid]; bi = s_ri[tid];
#pragma unroll
            for (int m = 16; m; m >>= 1) {
                float ov = __shfl_xor(v, m, 32);
                int   oi = __shfl_xor(bi, m, 32);
                if (ov > v || (ov == v && oi < bi)) { v = ov; bi = oi; }
            }
            if (tid == 0) { s_idx[s] = bi; s_last = bi; }
        }
        __syncthreads();
    }
    if (tid < NANCH) {
        int j = s_idx[tid];
        float* ap = anchors + (size_t)(g * NANCH + tid) * 3;
        ap[0] = pts[j * 4 + 0]; ap[1] = pts[j * 4 + 1]; ap[2] = pts[j * 4 + 2];
    }
}

// ---------------- ball-query + grouped feature + pos-emb -> tokens ----------------
// one block (256 thr) per anchor; 3 temporal offsets; wave0 does ballot compaction
__global__ void k_group(const float* __restrict__ radar, const float* __restrict__ anchors,
                        const float* __restrict__ Wd, const float* __restrict__ Wf,
                        const float* __restrict__ Wp, const float* __restrict__ bp,
                        float* __restrict__ X) {
    int aid = blockIdx.x;
    int g = aid / NANCH, a = aid % NANCH;
    int b = g / T_N, t = g % T_N;
    __shared__ float s_d2[NPTS];
    __shared__ int   s_sel[KNB];
    __shared__ float s_nx[KNB], s_ny[KNB], s_nz[KNB], s_nf[KNB];
    int tid = threadIdx.x;
    float ax = anchors[(size_t)aid * 3 + 0];
    float ay = anchors[(size_t)aid * 3 + 1];
    float az = anchors[(size_t)aid * 3 + 2];
    int d0 = tid, d1 = tid + 256;
    float wd00 = Wd[d0 * 4 + 0], wd01 = Wd[d0 * 4 + 1], wd02 = Wd[d0 * 4 + 2], wd03 = Wd[d0 * 4 + 3];
    float wd10 = Wd[d1 * 4 + 0], wd11 = Wd[d1 * 4 + 1], wd12 = Wd[d1 * 4 + 2], wd13 = Wd[d1 * 4 + 3];
    float wf0 = Wf[d0], wf1 = Wf[d1];
    float f0 = -3.4e38f, f1 = -3.4e38f;
    const float R2 = RADIUS_ * RADIUS_;

    for (int off = -1; off <= 1; ++off) {
        int pf = 2 * t + 1 + off;               // padded frame index
        int fr = (pf == 0) ? 0 : pf - 1;        // original frame (front-dup padding)
        const float* nb = radar + (size_t)((b * L_FR + fr) * NPTS) * 4;
        for (int p = tid; p < NPTS; p += 256) {
            float dx = nb[p * 4 + 0] - ax, dy = nb[p * 4 + 1] - ay, dz = nb[p * 4 + 2] - az;
            s_d2[p] = dx * dx + dy * dy + dz * dz;
        }
        __syncthreads();
        if (tid < 32) {
            int lane = tid, cnt = 0;
            for (int base = 0; base < NPTS && cnt < KNB; base += 32) {
                bool in = s_d2[base + lane] < R2;
                unsigned m = (unsigned)__ballot(in);
                int pos = cnt + __popc(m & ((1u << lane) - 1u));
                if (in && pos < KNB) s_sel[pos] = base + lane;
                cnt += __popc(m);
            }
            int total = cnt < KNB ? cnt : KNB;
            int v = (lane < total) ? s_sel[lane] : ((total > 0) ? s_sel[0] : 0);
            s_sel[lane] = v;
            const float* np_ = nb + (size_t)s_sel[lane] * 4;
            s_nx[lane] = np_[0]; s_ny[lane] = np_[1]; s_nz[lane] = np_[2]; s_nf[lane] = np_[3];
        }
        __syncthreads();
        float offf = (float)off;
        for (int k = 0; k < KNB; ++k) {
            float dx = s_nx[k] - ax, dy = s_ny[k] - ay, dz = s_nz[k] - az, gf = s_nf[k];
            f0 = fmaxf(f0, wd00 * dx + wd01 * dy + wd02 * dz + wd03 * offf + gf * wf0);
            f1 = fmaxf(f1, wd10 * dx + wd11 * dy + wd12 * dz + wd13 * offf + gf * wf1);
        }
        __syncthreads();
    }
    float tc = (float)(t + 1);
    float e0 = Wp[d0 * 4 + 0] * ax + Wp[d0 * 4 + 1] * ay + Wp[d0 * 4 + 2] * az + Wp[d0 * 4 + 3] * tc + bp[d0] + f0;
    float e1 = Wp[d1 * 4 + 0] * ax + Wp[d1 * 4 + 1] * ay + Wp[d1 * 4 + 2] * az + Wp[d1 * 4 + 3] * tc + bp[d1] + f1;
    size_t tok = (size_t)b * SEQ + (size_t)t * NANCH + a;
    X[tok * DIMC + d0] = e0;
    X[tok * DIMC + d1] = e1;
}

// ---------------- rmsnorm: one wave per token ----------------
__global__ void k_rmsnorm(const float* __restrict__ x, const float* __restrict__ w,
                          float* __restrict__ out) {
    int wv = threadIdx.x >> 5, lane = threadIdx.x & 31;
    size_t tok = (size_t)blockIdx.x * 8 + wv;
    const float* xp = x + tok * DIMC;
    float vals[16]; float s = 0.0f;
#pragma unroll
    for (int i = 0; i < 16; ++i) { float v = xp[lane + 32 * i]; vals[i] = v; s += v * v; }
#pragma unroll
    for (int m = 16; m; m >>= 1) s += __shfl_xor(s, m, 32);
    float r = rsqrtf(s / (float)DIMC + 1e-5f);
    float* op = out + tok * DIMC;
#pragma unroll
    for (int i = 0; i < 16; ++i) op[lane + 32 * i] = vals[i] * r * w[lane + 32 * i];
}

// ---------------- causal depthwise conv (width 4) + SiLU; xi = XR[:, :512] ----------------
__global__ void k_conv_silu(const float* __restrict__ XR, const float* __restrict__ cw,
                            const float* __restrict__ cb, float* __restrict__ XC) {
    int i = blockIdx.x * blockDim.x + threadIdx.x;
    if (i >= TOK * DIMC) return;
    int d = i & (DIMC - 1);
    int tg = i >> 9;
    int b = tg / SEQ, tt = tg % SEQ;
    float s = cb[d];
#pragma unroll
    for (int j = 0; j < DCONV; ++j) {
        int ts = tt - (DCONV - 1) + j;
        if (ts >= 0) s += cw[d * DCONV + j] * XR[(size_t)(b * SEQ + ts) * (2 * DIMC) + d];
    }
    XC[i] = s / (1.0f + __expf(-s));
}

__global__ void k_dt2bf(const float* __restrict__ XDBL, bf16_t* __restrict__ DTBF) {
    int i = blockIdx.x * blockDim.x + threadIdx.x;
    if (i >= TOK * DTRANK) return;
    int r = i >> 5, c = i & 31;
    DTBF[i] = (bf16_t)XDBL[(size_t)r * 64 + c];
}

__global__ void k_softplus(float* __restrict__ p, int n) {
    int i = blockIdx.x * blockDim.x + threadIdx.x;
    if (i >= n) return;
    float x = p[i];
    p[i] = (x > 20.0f) ? x : log1pf(__expf(x));
}

// ---------------- selective scan: grid = B, block = DIM; state in registers ----------------
__global__ void k_ssm(const float* __restrict__ DL, const float* __restrict__ XC,
                      const float* __restrict__ XDBL, const float* __restrict__ Alog,
                      const float* __restrict__ Dv, float* __restrict__ Y) {
    int b = blockIdx.x, d = threadIdx.x;
    float Ad[DSTATE], h[DSTATE];
#pragma unroll
    for (int n = 0; n < DSTATE; ++n) { Ad[n] = -__expf(Alog[d * DSTATE + n]); h[n] = 0.0f; }
    float dd = Dv[d];
    for (int t = 0; t < SEQ; ++t) {
        size_t row = (size_t)b * SEQ + t;
        float dl = DL[row * DIMC + d];
        float u  = XC[row * DIMC + d];
        const float* xd = XDBL + row * 64;
        float acc = 0.0f;
#pragma unroll
        for (int n = 0; n < DSTATE; ++n) {
            float dA = __expf(dl * Ad[n]);
            h[n] = dA * h[n] + dl * xd[32 + n] * u;
            acc += h[n] * xd[48 + n];
        }
        Y[row * DIMC + d] = acc + u * dd;
    }
}

// ---------------- gate: ABF = bf16( Y * silu(res) ), res = XR[:, 512:] ----------------
__global__ void k_gate(const float* __restrict__ Y, const float* __restrict__ XR,
                       bf16_t* __restrict__ ABF) {
    int i = blockIdx.x * blockDim.x + threadIdx.x;
    if (i >= TOK * DIMC) return;
    int col = i & (DIMC - 1);
    size_t row = (size_t)(i >> 9);
    float rv = XR[row * (2 * DIMC) + DIMC + col];
    float sil = rv / (1.0f + __expf(-rv));
    ABF[i] = (bf16_t)(Y[i] * sil);
}

// ---------------- final rmsnorm of last token per batch ----------------
__global__ void k_final_h(const float* __restrict__ X, const float* __restrict__ w,
                          float* __restrict__ H) {
    int b = blockIdx.x, lane = threadIdx.x;  // block = 32
    const float* xp = X + ((size_t)b * SEQ + (SEQ - 1)) * DIMC;
    float s = 0.0f;
    for (int i = lane; i < DIMC; i += 32) { float v = xp[i]; s += v * v; }
#pragma unroll
    for (int m = 16; m; m >>= 1) s += __shfl_xor(s, m, 32);
    float r = rsqrtf(s / (float)DIMC + 1e-5f);
    for (int i = lane; i < DIMC; i += 32) H[(size_t)b * DIMC + i] = xp[i] * r * w[i];
}

// ---------------- MLP head: layernorm -> relu(l1) -> l2 (+sigmoid) ----------------
__global__ void k_head(const float* __restrict__ H, const float* __restrict__ lnw,
                       const float* __restrict__ lnb, const float* __restrict__ W1,
                       const float* __restrict__ b1, const float* __restrict__ W2,
                       const float* __restrict__ b2, int hidden, int outdim,
                       float* __restrict__ out, int use_sig) {
    int b = blockIdx.x, tid = threadIdx.x;  // block = 128
    __shared__ float s_ln[DIMC];
    __shared__ float s_hid[128];
    __shared__ float s_red[128];
    const float* hp = H + (size_t)b * DIMC;
    float l0 = hp[tid], l1 = hp[tid + 128], l2 = hp[tid + 256], l3 = hp[tid + 384];
    s_red[tid] = l0 + l1 + l2 + l3;
    __syncthreads();
    for (int st = 64; st > 0; st >>= 1) { if (tid < st) s_red[tid] += s_red[tid + st]; __syncthreads(); }
    float mean = s_red[0] / (float)DIMC;
    __syncthreads();
    float e0 = l0 - mean, e1 = l1 - mean, e2 = l2 - mean, e3 = l3 - mean;
    s_red[tid] = e0 * e0 + e1 * e1 + e2 * e2 + e3 * e3;
    __syncthreads();
    for (int st = 64; st > 0; st >>= 1) { if (tid < st) s_red[tid] += s_red[tid + st]; __syncthreads(); }
    float rv = rsqrtf(s_red[0] / (float)DIMC + 1e-5f);
    s_ln[tid]       = e0 * rv * lnw[tid]       + lnb[tid];
    s_ln[tid + 128] = e1 * rv * lnw[tid + 128] + lnb[tid + 128];
    s_ln[tid + 256] = e2 * rv * lnw[tid + 256] + lnb[tid + 256];
    s_ln[tid + 384] = e3 * rv * lnw[tid + 384] + lnb[tid + 384];
    __syncthreads();
    for (int j = tid; j < hidden; j += 128) {
        float s = b1[j];
        for (int k = 0; k < DIMC; ++k) s += s_ln[k] * W1[(size_t)j * DIMC + k];
        s_hid[j] = fmaxf(s, 0.0f);
    }
    __syncthreads();
    for (int o = tid; o < outdim; o += 128) {
        float s = b2[o];
        for (int k = 0; k < hidden; ++k) s += s_hid[k] * W2[(size_t)o * hidden + k];
        if (use_sig) s = 1.0f / (1.0f + __expf(-s));
        out[(size_t)b * outdim + o] = s;
    }
}

// ---------------- launch ----------------
extern "C" void kernel_launch(void* const* d_in, const int* in_sizes, int n_in,
                              void* d_out, int out_size, void* d_ws, size_t ws_size,
                              hipStream_t stream) {
    (void)in_sizes; (void)n_in; (void)out_size; (void)ws_size;
    auto F = [&](int i) { return (const float*)d_in[i]; };
    const float* radar = F(P_RADAR);

    char* ws = (char*)d_ws;
    size_t off = 0;
    auto alloc = [&](size_t bytes) { size_t o = off; off += (bytes + 255) & ~(size_t)255; return o; };
    float*  X    = (float*)(ws + alloc((size_t)TOK * DIMC * 4));      // residual stream / token emb
    float*  XN   = (float*)(ws + alloc((size_t)TOK * DIMC * 4));      // rmsnorm out
    float*  XR   = (float*)(ws + alloc((size_t)TOK * 2 * DIMC * 4));  // in_proj out (xi | res)
    float*  XC   = (float*)(ws + alloc((size_t)TOK * DIMC * 4));      // conv+silu out (u)
    float*  DLb  = (float*)(ws + alloc((size_t)TOK * DIMC * 4));      // delta
    float*  Yb   = (float*)(ws + alloc((size_t)TOK * DIMC * 4));      // ssm out
    float*  XDBL = (float*)(ws + alloc((size_t)TOK * 64 * 4));        // x_proj out (dt|B|C)
    float*  ANCH = (float*)(ws + alloc((size_t)GRP * NANCH * 3 * 4));
    float*  HB   = (float*)(ws + alloc((size_t)B_N * DIMC * 4));
    bf16_t* ABF  = (bf16_t*)(ws + alloc((size_t)TOK * DIMC * 2));     // bf16 activations
    bf16_t* WBF  = (bf16_t*)(ws + alloc((size_t)1024 * 512 * 2));     // bf16 weights (reused)
    bf16_t* DTBF = (bf16_t*)(ws + alloc((size_t)TOK * DTRANK * 2));

    const int EL = TOK * DIMC;                  // 2,097,152
    const int EB = (EL + 255) / 256;

    // stage 1: FPS anchors, grouped features + positional embedding -> X
    k_fps<<<GRP, 1024, 0, stream>>>(radar, ANCH);
    k_group<<<GRP * NANCH, 256, 0, stream>>>(radar, ANCH, F(P_CONV_D), F(P_CONV_F),
                                             F(P_POS_W), F(P_POS_B), X);

    // stage 2: Mamba layers
    for (int l = 0; l < DEPTH; ++l) {
        int base = P_LAYER0 + l * 10;
        const float* Alog = F(base + 0); const float* Dv  = F(base + 1);
        const float* cb   = F(base + 2); const float* cwt = F(base + 3);
        const float* dtW  = F(base + 4); const float* dtb = F(base + 5);
        const float* inW  = F(base + 6); const float* nw  = F(base + 7);
        const float* outW = F(base + 8); const float* xW  = F(base + 9);

        k_rmsnorm<<<TOK / 8, 256, 0, stream>>>(X, nw, XN);
        k_f2bf<<<EB, 256, 0, stream>>>(XN, ABF, EL);
        k_f2bf<<<(1024 * 512 + 255) / 256, 256, 0, stream>>>(inW, WBF, 1024 * 512);
        k_gemm<<<dim3(1024 / 64, TOK / 128), 256, 0, stream>>>(ABF, WBF, nullptr, nullptr, XR, TOK, 1024, 512);

        k_conv_silu<<<EB, 256, 0, stream>>>(XR, cwt, cb, XC);

        k_f2bf<<<EB, 256, 0, stream>>>(XC, ABF, EL);
        k_f2bf<<<(64 * 512 + 255) / 256, 256, 0, stream>>>(xW, WBF, 64 * 512);
        k_gemm<<<dim3(1, TOK / 128), 256, 0, stream>>>(ABF, WBF, nullptr, nullptr, XDBL, TOK, 64, 512);

        k_dt2bf<<<(TOK * DTRANK + 255) / 256, 256, 0, stream>>>(XDBL, DTBF);
        k_f2bf<<<(512 * 32 + 255) / 256, 256, 0, stream>>>(dtW, WBF, 512 * 32);
        k_gemm<<<dim3(512 / 64, TOK / 128), 256, 0, stream>>>(DTBF, WBF, dtb, nullptr, DLb, TOK, 512, 32);
        k_softplus<<<EB, 256, 0, stream>>>(DLb, EL);

        k_ssm<<<B_N, DIMC, 0, stream>>>(DLb, XC, XDBL, Alog, Dv, Yb);

        k_gate<<<EB, 256, 0, stream>>>(Yb, XR, ABF);
        k_f2bf<<<(512 * 512 + 255) / 256, 256, 0, stream>>>(outW, WBF, 512 * 512);
        k_gemm<<<dim3(512 / 64, TOK / 128), 256, 0, stream>>>(ABF, WBF, nullptr, X, X, TOK, 512, 512);
    }

    // stage 3: final norm of last token + heads
    k_final_h<<<B_N, 32, 0, stream>>>(X, F(P_NORM_F), HB);
    float* out = (float*)d_out;
    // out layout: betas(4x10)@0, pose(4x63)@40, root(4x3)@292, trans(4x3)@304, gender(4x1)@316
    k_head<<<B_N, 128, 0, stream>>>(HB, F(P_BETAS + 5), F(P_BETAS + 4), F(P_BETAS + 0), F(P_BETAS + 1),
                                    F(P_BETAS + 2), F(P_BETAS + 3), 128, 10, out + 0, 0);
    k_head<<<B_N, 128, 0, stream>>>(HB, F(P_POSE + 5), F(P_POSE + 4), F(P_POSE + 0), F(P_POSE + 1),
                                    F(P_POSE + 2), F(P_POSE + 3), 128, 63, out + 40, 0);
    k_head<<<B_N, 128, 0, stream>>>(HB, F(P_ROOT + 5), F(P_ROOT + 4), F(P_ROOT + 0), F(P_ROOT + 1),
                                    F(P_ROOT + 2), F(P_ROOT + 3), 128, 3, out + 292, 0);
    k_head<<<B_N, 128, 0, stream>>>(HB, F(P_TRANS + 5), F(P_TRANS + 4), F(P_TRANS + 0), F(P_TRANS + 1),
                                    F(P_TRANS + 2), F(P_TRANS + 3), 128, 3, out + 304, 0);
    k_head<<<B_N, 128, 0, stream>>>(HB, F(P_GENDER + 5), F(P_GENDER + 4), F(P_GENDER + 0), F(P_GENDER + 1),
                                    F(P_GENDER + 2), F(P_GENDER + 3), 64, 1, out + 316, 1);
}
